// GroupedQueryAttention_6004364280097
// MI455X (gfx1250) — matrix operand order repaired and checked by reading, not compiled, not run on hardware
//
#include <hip/hip_runtime.h>
#include <hip/hip_bf16.h>
#include <math.h>

// Problem constants (B,T,E)=(2,2048,1024), HQ=16, HKV=4, D=64, G=4
#define Bn   2
#define Tn   2048
#define En   1024
#define HQn  16
#define HKVn 4
#define Dn   64
#define Gn   4

typedef __attribute__((ext_vector_type(16))) __bf16 v16bf;
typedef __attribute__((ext_vector_type(8)))  float  v8f;

union FragBF {
  v16bf v;
  unsigned u[8];
  uint4    q[2];
};

__device__ __forceinline__ unsigned short f32_to_bf16(float f) {
  unsigned u = __float_as_uint(f);
  unsigned r = (u + 0x7FFFu + ((u >> 16) & 1u)) >> 16;  // round-to-nearest-even
  return (unsigned short)r;
}

// ---------------- elementwise fp32 -> bf16 ----------------
__global__ void cvt_f32_bf16(const float* __restrict__ src, __bf16* __restrict__ dst, int n) {
  int i = blockIdx.x * blockDim.x + threadIdx.x;
  if (i < n) ((unsigned short*)dst)[i] = f32_to_bf16(src[i]);
}

// ---------------- generic WMMA GEMM: C(MxN,f32) = A(MxK,bf16) @ B(KxN,bf16) + bias ----------------
// One wave32 computes a 16x16 tile via v_wmma_f32_16x16x32_bf16. All operands are L2-resident.
__global__ void wmma_gemm_bf16(const __bf16* __restrict__ A, const __bf16* __restrict__ Bw,
                               const float* __restrict__ bias, float* __restrict__ C,
                               int M, int N, int K) {
  const int lane = threadIdx.x & 31;
  const int wave = threadIdx.x >> 5;
  const int tile = blockIdx.x * (blockDim.x >> 5) + wave;
  const int ntiles = N >> 4;
  const int mt = tile / ntiles;
  const int nt = tile - mt * ntiles;
  if (mt * 16 >= M) return;
  const int h  = lane >> 4;     // half-wave select
  const int nl = lane & 15;
  const int m  = mt * 16 + nl;  // A-fragment row owned by this lane
  const int n0 = nt * 16;
  const unsigned* arow = (const unsigned*)(A + (size_t)m * K);
  v8f acc = {};
  for (int k0 = 0; k0 < K; k0 += 32) {
    FragBF a, bf;
#pragma unroll
    for (int v2 = 0; v2 < 8; ++v2) {
      int kk = k0 + ((v2 >> 2) << 4) + (h << 3) + ((v2 & 3) << 1);
      a.u[v2] = arow[kk >> 1];
    }
    const unsigned short* bcol = (const unsigned short*)Bw + n0 + nl;
#pragma unroll
    for (int v2 = 0; v2 < 8; ++v2) {
      int kB = k0 + ((v2 >> 2) << 4) + (h << 3) + ((v2 & 3) << 1);
      bf.u[v2] = (unsigned)bcol[(size_t)kB * N] | ((unsigned)bcol[(size_t)(kB + 1) * N] << 16);
    }
    acc = __builtin_amdgcn_wmma_f32_16x16x32_bf16(false, a.v, false, bf.v,
                                                  (short)0, acc, false, false);
  }
  const int n = n0 + nl;
  const float bb = bias[n];
#pragma unroll
  for (int r = 0; r < 8; ++r) {
    const int mm = mt * 16 + r + (h << 3);
    C[(size_t)mm * N + n] = acc[r] + bb;
  }
}

// ---------------- RoPE on Q, pack to bf16 (B,HQ,T,D) ----------------
__global__ void rope_q_pack(const float* __restrict__ Qt, __bf16* __restrict__ Qb) {
  int idx = blockIdx.x * blockDim.x + threadIdx.x;  // B*T*HQ*32 threads
  int j  = idx & 31;
  int hq = (idx >> 5) & 15;
  int t  = (idx >> 9) & 2047;
  int b  = idx >> 20;
  float theta = __expf(-(float)j * 0.28782313662425572f);   // ln(10000)/32
  float ang = (float)(t + 1) * theta;
  float sn, cs;
  __sincosf(ang, &sn, &cs);
  const float* src = Qt + ((size_t)(b * Tn + t)) * En + hq * Dn;
  float x1 = src[j], x2 = src[j + 32];
  unsigned short* dst = (unsigned short*)Qb + ((size_t)(b * HQn + hq) * Tn + t) * Dn;
  dst[j]      = f32_to_bf16(x1 * cs - x2 * sn);
  dst[j + 32] = f32_to_bf16(x2 * cs + x1 * sn);
}

// ---------------- RoPE on K, pack to bf16 TRANSPOSED (B,HKV,D,T) ----------------
__global__ void rope_k_pack(const float* __restrict__ Kt, __bf16* __restrict__ Ktb) {
  int idx = blockIdx.x * blockDim.x + threadIdx.x;  // B*T*HKV*32 threads
  int j  = idx & 31;
  int hk = (idx >> 5) & 3;
  int t  = (idx >> 7) & 2047;
  int b  = idx >> 18;
  float theta = __expf(-(float)j * 0.28782313662425572f);
  float ang = (float)(t + 1) * theta;
  float sn, cs;
  __sincosf(ang, &sn, &cs);
  const float* src = Kt + ((size_t)(b * Tn + t)) * (HKVn * Dn) + hk * Dn;
  float x1 = src[j], x2 = src[j + 32];
  unsigned short* dst = (unsigned short*)Ktb + ((size_t)(b * HKVn + hk) * Dn) * Tn;
  dst[(size_t)j * Tn + t]        = f32_to_bf16(x1 * cs - x2 * sn);
  dst[(size_t)(j + 32) * Tn + t] = f32_to_bf16(x2 * cs + x1 * sn);
}

// ---------------- V pack to bf16 (B,HKV,T,D) ----------------
__global__ void v_pack(const float* __restrict__ Vt, __bf16* __restrict__ Vb) {
  int idx = blockIdx.x * blockDim.x + threadIdx.x;  // B*T*HKV*D threads
  int d  = idx & 63;
  int hk = (idx >> 6) & 3;
  int t  = (idx >> 8) & 2047;
  int b  = idx >> 19;
  float v = Vt[((size_t)(b * Tn + t)) * (HKVn * Dn) + hk * Dn + d];
  ((unsigned short*)Vb)[((size_t)(b * HKVn + hk) * Tn + t) * Dn + d] = f32_to_bf16(v);
}

// Score tile for 16 q-rows x 16 k-cols: 2 WMMAs over D=64 (raw, unscaled).
__device__ __forceinline__ v8f score_tile(const FragBF& qa0, const FragBF& qa1,
                                          const unsigned short* khead, int tk0, int lane) {
  const int h = lane >> 4, nl = lane & 15;
  FragBF b0, b1;
  const unsigned short* bc = khead + tk0 + nl;
#pragma unroll
  for (int v2 = 0; v2 < 8; ++v2) {
    int kk = ((v2 >> 2) << 4) + (h << 3) + ((v2 & 3) << 1);
    b0.u[v2] = (unsigned)bc[(size_t)kk * Tn] | ((unsigned)bc[(size_t)(kk + 1) * Tn] << 16);
    b1.u[v2] = (unsigned)bc[(size_t)(kk + 32) * Tn] | ((unsigned)bc[(size_t)(kk + 33) * Tn] << 16);
  }
  v8f acc = {};
  acc = __builtin_amdgcn_wmma_f32_16x16x32_bf16(false, qa0.v, false, b0.v, (short)0, acc, false, false);
  acc = __builtin_amdgcn_wmma_f32_16x16x32_bf16(false, qa1.v, false, b1.v, (short)0, acc, false, false);
  return acc;
}

// ---------------- fused attention: scores + online softmax + att_weights write + P@V ----------------
// One wave32 owns a 16-token q-tile. Pass 1 streams K-tiles and keeps per-row running
// (max, sumexp) online; pass 2 recomputes scores (cheap), writes normalized probabilities
// to att exactly once (NON-TEMPORAL: write-once data must not evict L2-resident K/V/Q),
// transposes the P tile C-layout -> A-layout through LDS, and accumulates P@V with
// 4 WMMAs per 32-token k-step.
__global__ void attn_fused(const __bf16* __restrict__ Qb, const __bf16* __restrict__ Ktb,
                           const __bf16* __restrict__ Vb, float* __restrict__ att,
                           __bf16* __restrict__ ybf) {
  __shared__ unsigned short ptile[4][16 * 32];  // per-wave 16x32 bf16 staging (1 KB/wave)
  const int lane = threadIdx.x & 31;
  const int wave = threadIdx.x >> 5;
  const int tqt = blockIdx.x * 4 + wave;  // 0..127
  const int bh  = blockIdx.y;             // 0..31
  const int b = bh >> 4, hq = bh & 15;
  const int hkv = hq >> 2, g = hq & 3;    // hq = hkv*G + g
  const int tq0 = tqt << 4;
  const int h = lane >> 4, nl = lane & 15;
  unsigned short* lds = ptile[wave];

  // Q A-fragments (k = 0..31 and 32..63), loaded once
  const unsigned* qrow = (const unsigned*)((const unsigned short*)Qb +
                                           ((size_t)(b * HQn + hq) * Tn + tq0 + nl) * Dn);
  FragBF qa0, qa1;
#pragma unroll
  for (int v2 = 0; v2 < 8; ++v2) {
    int kk = ((v2 >> 2) << 4) + (h << 3) + ((v2 & 3) << 1);
    qa0.u[v2] = qrow[kk >> 1];
    qa1.u[v2] = qrow[(kk + 32) >> 1];
  }
  const unsigned short* khead = (const unsigned short*)Ktb + (size_t)(b * HKVn + hkv) * Dn * Tn;
  const unsigned short* vhead = (const unsigned short*)Vb + (size_t)(b * HKVn + hkv) * Tn * Dn;
  float* rowBase = att + ((size_t)((b * Gn + g) * HKVn + hkv)) * Tn * Tn;

  // ---- pass 1: online (max, sumexp) per row ----
  float mrow[8], lrow[8];
#pragma unroll
  for (int r = 0; r < 8; ++r) { mrow[r] = -1e30f; lrow[r] = 0.f; }
  for (int tkt = 0; tkt <= tqt; ++tkt) {
    const int tk0 = tkt << 4;
    const int tk = tk0 + nl;
    v8f s = score_tile(qa0, qa1, khead, tk0, lane);
#pragma unroll
    for (int r = 0; r < 8; ++r) {
      const int tq = tq0 + r + (h << 3);
      float sv = s[r] * 0.125f;            // 1/sqrt(64)
      if (tk > tq) sv = -1e30f;            // causal (finite sentinel: no NaN in rescale)
      const float mn = fmaxf(mrow[r], sv);
      lrow[r] = lrow[r] * __expf(mrow[r] - mn) + __expf(sv - mn);
      mrow[r] = mn;
    }
  }
  // cross-lane combine across the 16 lanes that share each row
#pragma unroll
  for (int d = 1; d < 16; d <<= 1) {
#pragma unroll
    for (int r = 0; r < 8; ++r) {
      const float mo = __shfl_xor(mrow[r], d, 32);
      const float lo = __shfl_xor(lrow[r], d, 32);
      const float mn = fmaxf(mrow[r], mo);
      lrow[r] = lrow[r] * __expf(mrow[r] - mn) + lo * __expf(mo - mn);
      mrow[r] = mn;
    }
  }
  float inv[8];
#pragma unroll
  for (int r = 0; r < 8; ++r) inv[r] = 1.0f / lrow[r];

  // ---- pass 2: recompute, normalize, write att once (NT), accumulate P@V ----
  v8f acc0 = {}, acc1 = {}, acc2 = {}, acc3 = {};
  const int ksteps = (tqt >> 1) + 1;  // 32-token k-steps covering tk <= tq0+15
  for (int j = 0; j < ksteps; ++j) {
#pragma unroll
    for (int half = 0; half < 2; ++half) {
      const int tkt = 2 * j + half;
      const int tk0 = tkt << 4;
      const int tk = tk0 + nl;
      if (tkt <= tqt) {
        v8f s = score_tile(qa0, qa1, khead, tk0, lane);
#pragma unroll
        for (int r = 0; r < 8; ++r) {
          const int tq = tq0 + r + (h << 3);
          float p = __expf(s[r] * 0.125f - mrow[r]) * inv[r];
          if (tk > tq) p = 0.f;            // exact zero like softmax(-inf)
          __builtin_nontemporal_store(p, &rowBase[(size_t)tq * Tn + tk]);
          lds[(r + (h << 3)) * 32 + nl + (half << 4)] = f32_to_bf16(p);
        }
      } else {                             // odd boundary: pad second half with zeros
#pragma unroll
        for (int r = 0; r < 8; ++r) {
          __builtin_nontemporal_store(0.f, &rowBase[(size_t)(tq0 + r + (h << 3)) * Tn + tk]);
          lds[(r + (h << 3)) * 32 + nl + (half << 4)] = 0;
        }
      }
    }
    // P tile: C-layout -> A-fragment via LDS (same-wave DS ops are in order)
    FragBF a;
    const unsigned* prow = (const unsigned*)(lds + nl * 32);
#pragma unroll
    for (int v2 = 0; v2 < 8; ++v2) {
      int kk = ((v2 >> 2) << 4) + (h << 3) + ((v2 & 3) << 1);
      a.u[v2] = prow[kk >> 1];
    }
    const unsigned short* vcol = vhead + (size_t)(j << 5) * Dn + nl;
    FragBF b0, b1, b2, b3;
#pragma unroll
    for (int v2 = 0; v2 < 8; ++v2) {
      int kk = ((v2 >> 2) << 4) + (h << 3) + ((v2 & 3) << 1);
      const unsigned short* p0 = vcol + (size_t)kk * Dn;
      const unsigned short* p1 = p0 + Dn;
      b0.u[v2] = (unsigned)p0[0]  | ((unsigned)p1[0]  << 16);
      b1.u[v2] = (unsigned)p0[16] | ((unsigned)p1[16] << 16);
      b2.u[v2] = (unsigned)p0[32] | ((unsigned)p1[32] << 16);
      b3.u[v2] = (unsigned)p0[48] | ((unsigned)p1[48] << 16);
    }
    acc0 = __builtin_amdgcn_wmma_f32_16x16x32_bf16(false, a.v, false, b0.v, (short)0, acc0, false, false);
    acc1 = __builtin_amdgcn_wmma_f32_16x16x32_bf16(false, a.v, false, b1.v, (short)0, acc1, false, false);
    acc2 = __builtin_amdgcn_wmma_f32_16x16x32_bf16(false, a.v, false, b2.v, (short)0, acc2, false, false);
    acc3 = __builtin_amdgcn_wmma_f32_16x16x32_bf16(false, a.v, false, b3.v, (short)0, acc3, false, false);
  }

  // zero-fill the remaining masked att region (softmax of -inf == 0), non-temporal
  for (int tkt = 2 * ksteps; tkt < (Tn >> 4); ++tkt) {
    const int tk = (tkt << 4) + nl;
#pragma unroll
    for (int r = 0; r < 8; ++r)
      __builtin_nontemporal_store(0.f, &rowBase[(size_t)(tq0 + r + (h << 3)) * Tn + tk]);
  }

  // store y tile as bf16 (B,T,E) for the output projection (regular temporal: read back next)
  unsigned short* ybase = (unsigned short*)ybf;
#pragma unroll
  for (int r = 0; r < 8; ++r) {
    const int t = tq0 + r + (h << 3);
    unsigned short* yrow = ybase + ((size_t)(b * Tn + t)) * En + hq * Dn;
    yrow[nl]      = f32_to_bf16(acc0[r]);
    yrow[16 + nl] = f32_to_bf16(acc1[r]);
    yrow[32 + nl] = f32_to_bf16(acc2[r]);
    yrow[48 + nl] = f32_to_bf16(acc3[r]);
  }
}

extern "C" void kernel_launch(void* const* d_in, const int* in_sizes, int n_in,
                              void* d_out, int out_size, void* d_ws, size_t ws_size,
                              hipStream_t stream) {
  const float* x  = (const float*)d_in[0];
  // d_in[1] = mask: pure causal triu, applied analytically in attn_fused
  const float* Wq = (const float*)d_in[2];
  const float* bq = (const float*)d_in[3];
  const float* Wk = (const float*)d_in[4];
  const float* bk = (const float*)d_in[5];
  const float* Wv = (const float*)d_in[6];
  const float* bv = (const float*)d_in[7];
  const float* Wo = (const float*)d_in[8];
  const float* bo = (const float*)d_in[9];

  float* y_out = (float*)d_out;                         // (B,T,E) fp32
  float* att   = (float*)d_out + (size_t)Bn * Tn * En;  // (B,G,HKV,T,T) fp32

  char* ws = (char*)d_ws;
  size_t off = 0;
  auto wsalloc = [&](size_t bytes) -> void* {
    void* p = ws + off;
    off += (bytes + 255) & ~(size_t)255;
    return p;
  };
  const size_t NTOK = (size_t)Bn * Tn;                      // 4096
  __bf16* xh   = (__bf16*)wsalloc(NTOK * En * 2);           // 8 MB
  __bf16* Wqb  = (__bf16*)wsalloc((size_t)En * En * 2);     // 2 MB
  __bf16* Wkb  = (__bf16*)wsalloc((size_t)En * HKVn * Dn * 2);
  __bf16* Wvb  = (__bf16*)wsalloc((size_t)En * HKVn * Dn * 2);
  __bf16* Wob  = (__bf16*)wsalloc((size_t)En * En * 2);
  float*  Qtmp = (float*)wsalloc(NTOK * En * 4);            // 16 MB
  float*  Ktmp = (float*)wsalloc(NTOK * HKVn * Dn * 4);     // 4 MB
  float*  Vtmp = (float*)wsalloc(NTOK * HKVn * Dn * 4);     // 4 MB
  __bf16* Qb   = (__bf16*)wsalloc(NTOK * En * 2);           // 8 MB  (B,HQ,T,D)
  __bf16* Ktb  = (__bf16*)wsalloc(NTOK * HKVn * Dn * 2);    // 2 MB  (B,HKV,D,T)
  __bf16* Vbb  = (__bf16*)wsalloc(NTOK * HKVn * Dn * 2);    // 2 MB  (B,HKV,T,D)
  __bf16* ybf  = (__bf16*)wsalloc(NTOK * En * 2);           // 8 MB  (B,T,E)

  // 1) bf16 conversions
  cvt_f32_bf16<<<16384, 256, 0, stream>>>(x,  xh,  (int)(NTOK * En));
  cvt_f32_bf16<<<4096,  256, 0, stream>>>(Wq, Wqb, En * En);
  cvt_f32_bf16<<<1024,  256, 0, stream>>>(Wk, Wkb, En * HKVn * Dn);
  cvt_f32_bf16<<<1024,  256, 0, stream>>>(Wv, Wvb, En * HKVn * Dn);
  cvt_f32_bf16<<<4096,  256, 0, stream>>>(Wo, Wob, En * En);

  // 2) QKV projections (WMMA): M=4096, K=1024
  wmma_gemm_bf16<<<4096, 128, 0, stream>>>(xh, Wqb, bq, Qtmp, (int)NTOK, En, En);
  wmma_gemm_bf16<<<1024, 128, 0, stream>>>(xh, Wkb, bk, Ktmp, (int)NTOK, HKVn * Dn, En);
  wmma_gemm_bf16<<<1024, 128, 0, stream>>>(xh, Wvb, bv, Vtmp, (int)NTOK, HKVn * Dn, En);

  // 3) RoPE + head-layout packs
  rope_q_pack<<<8192, 256, 0, stream>>>(Qtmp, Qb);
  rope_k_pack<<<2048, 256, 0, stream>>>(Ktmp, Ktb);
  v_pack<<<8192, 256, 0, stream>>>(Vtmp, Vbb);

  // 4) fused attention: scores + online softmax + att write (once, NT) + P@V
  attn_fused<<<dim3(32, 32), 128, 0, stream>>>(Qb, Ktb, Vbb, att, ybf);

  // 5) output projection
  wmma_gemm_bf16<<<4096, 128, 0, stream>>>(ybf, Wob, bo, y_out, (int)NTOK, En, En);
}
